// GaussianAnsatzNN_57999238365632
// MI455X (gfx1250) — matrix operand order repaired and checked by reading, not compiled, not run on hardware
//
#include <hip/hip_runtime.h>
#include <hip/hip_bf16.h>

typedef __attribute__((ext_vector_type(16))) _Float16 v16h;
typedef __attribute__((ext_vector_type(8)))  float    v8f;

#define LOG2E      1.4426950408889634f
#define NEGH_LOG2E 0.7213475204444817f   // 0.5 * log2(e)

// ---------------------------------------------------------------------------
// Prep: transpose means, precompute exponent constants, fold norm into theta
//   bh  : [5][M] f16  -> B rows (mu_x, mu_y, mu_z, 1, 0); row 4 feeds the 12
//                        zero columns of the WMMA B operand (branchless load)
//   mt  : [3][M] f32  -> means transposed
//   q   : [M]    f32  -> -0.5*log2e*||mu||^2   (exponent bias per center)
//   ths : [M]    f32  -> theta * (2*pi)^{-3/2}
// ---------------------------------------------------------------------------
__global__ void gauss_prep_kernel(const float* __restrict__ means,
                                  const float* __restrict__ theta,
                                  _Float16* __restrict__ bh,
                                  float* __restrict__ mt,
                                  float* __restrict__ q,
                                  float* __restrict__ ths,
                                  int M)
{
    int m = blockIdx.x * blockDim.x + threadIdx.x;
    if (m >= M) return;
    float m0 = means[3*m+0], m1 = means[3*m+1], m2 = means[3*m+2];
    bh[0*M + m] = (_Float16)m0;
    bh[1*M + m] = (_Float16)m1;
    bh[2*M + m] = (_Float16)m2;
    bh[3*M + m] = (_Float16)1.0f;
    bh[4*M + m] = (_Float16)0.0f;       // shared zero row for columns 4..15
    mt[0*M + m] = m0;
    mt[1*M + m] = m1;
    mt[2*M + m] = m2;
    q[m]   = -NEGH_LOG2E * (m0*m0 + m1*m1 + m2*m2);
    ths[m] = theta[m] * 0.06349363593424097f;   // (2*pi)^{-3/2}, sigma = 1
}

// ---------------------------------------------------------------------------
// Per-chunk operand builder: exponent = xs.mu + q[m] + rr (already in log2
// domain) -> v_exp_f32 -> *theta -> f16 A fragment; plus the lane's 32B B
// fragment (row pointer selects data row 0-3 or the zero row).
// ---------------------------------------------------------------------------
__device__ __forceinline__ void gauss_chunk(
    int chunk, int hi,
    const float* __restrict__ mtx, const float* __restrict__ mty,
    const float* __restrict__ mtz, const float* __restrict__ q,
    const float* __restrict__ ths,
    const _Float16* __restrict__ bcol,
    float xs0, float xs1, float xs2, float rr,
    v16h& a, v16h& b)
{
    const int b1 = chunk + (hi << 3);   // lanes 0-15: chunk, lanes 16-31: +8

    v8f mx_lo = *(const v8f*)(mtx + b1);
    v8f mx_hi = *(const v8f*)(mtx + b1 + 16);
    v8f my_lo = *(const v8f*)(mty + b1);
    v8f my_hi = *(const v8f*)(mty + b1 + 16);
    v8f mz_lo = *(const v8f*)(mtz + b1);
    v8f mz_hi = *(const v8f*)(mtz + b1 + 16);
    v8f q_lo  = *(const v8f*)(q   + b1);
    v8f q_hi  = *(const v8f*)(q   + b1 + 16);
    v8f th_lo = *(const v8f*)(ths + b1);
    v8f th_hi = *(const v8f*)(ths + b1 + 16);

    #pragma unroll
    for (int j = 0; j < 8; ++j) {
        float arg = fmaf(xs0, mx_lo[j],
                    fmaf(xs1, my_lo[j],
                    fmaf(xs2, mz_lo[j], q_lo[j] + rr)));
        a[j] = (_Float16)(__builtin_amdgcn_exp2f(arg) * th_lo[j]);

        arg = fmaf(xs0, mx_hi[j],
              fmaf(xs1, my_hi[j],
              fmaf(xs2, mz_hi[j], q_hi[j] + rr)));
        a[8 + j] = (_Float16)(__builtin_amdgcn_exp2f(arg) * th_hi[j]);
    }

    // Unconditional 2x16B load: K = hi*8 .. hi*8+7 and 16+hi*8 .. 16+hi*8+7 of column `col`.
    typedef __attribute__((ext_vector_type(8))) _Float16 v8h_t;
    v8h_t b_lo = *(const v8h_t*)(bcol + b1);
    v8h_t b_hi = *(const v8h_t*)(bcol + b1 + 16);
    b = __builtin_shufflevector(b_lo, b_hi, 0, 1, 2, 3, 4, 5, 6, 7, 8, 9, 10, 11, 12, 13, 14, 15);
}

// ---------------------------------------------------------------------------
// Main kernel: one wave per 16-row tile of x. Two independent WMMA
// accumulation chains (unroll-2) hide the f16-WMMA RAW/co-exec hazard.
// Epilogue: out[k,d] = S_d - x[k,d] * S_3  (S_3 broadcast via wave32 shuffle).
// ---------------------------------------------------------------------------
__global__ void __launch_bounds__(256)
gauss_grad_wmma_kernel(const float* __restrict__ x,
                       const _Float16* __restrict__ bh,
                       const float* __restrict__ mt,
                       const float* __restrict__ q,
                       const float* __restrict__ ths,
                       float* __restrict__ out,
                       int K, int M)
{
    const int lane = threadIdx.x & 31;
    const int wave = threadIdx.x >> 5;
    const int tile = blockIdx.x * 8 + wave;
    if (tile * 16 >= K) return;          // wave-uniform: EXEC stays all-ones

    const int hi  = lane >> 4;           // 0: lanes 0-15, 1: lanes 16-31
    const int col = lane & 15;           // B/C column this lane owns

    // A-matrix row for this lane (both lane halves cover M=0..15)
    const int xrow = tile * 16 + col;
    const float x0 = x[3*xrow + 0];
    const float x1 = x[3*xrow + 1];
    const float x2 = x[3*xrow + 2];

    // Pre-scale into the log2 domain:
    //   exponent = (log2e*x).mu  -  0.5*log2e*||x||^2  -  0.5*log2e*||mu||^2
    const float xs0 = x0 * LOG2E;
    const float xs1 = x1 * LOG2E;
    const float xs2 = x2 * LOG2E;
    const float rr  = -NEGH_LOG2E * (x0*x0 + x1*x1 + x2*x2);

    const float* mtx = mt;
    const float* mty = mt + M;
    const float* mtz = mt + 2*M;

    // Columns 0-3 read (mu_x|mu_y|mu_z|1); columns 4-15 read the zero row.
    const int brow = (col < 4) ? col : 4;
    const _Float16* bcol = bh + (size_t)brow * M;

    v8f c0 = {};
    v8f c1 = {};

    int chunk = 0;
    for (; chunk + 64 <= M; chunk += 64) {
        v16h a0, b0, a1, b1;
        gauss_chunk(chunk,      hi, mtx, mty, mtz, q, ths, bcol,
                    xs0, xs1, xs2, rr, a0, b0);
        gauss_chunk(chunk + 32, hi, mtx, mty, mtz, q, ths, bcol,
                    xs0, xs1, xs2, rr, a1, b1);
        c0 = __builtin_amdgcn_wmma_f32_16x16x32_f16(
                false, a0, false, b0, (short)0, c0, false, false);
        c1 = __builtin_amdgcn_wmma_f32_16x16x32_f16(
                false, a1, false, b1, (short)0, c1, false, false);
    }
    for (; chunk < M; chunk += 32) {     // tail (M multiple of 32, not 64)
        v16h a0, b0;
        gauss_chunk(chunk, hi, mtx, mty, mtz, q, ths, bcol,
                    xs0, xs1, xs2, rr, a0, b0);
        c0 = __builtin_amdgcn_wmma_f32_16x16x32_f16(
                false, a0, false, b0, (short)0, c0, false, false);
    }

    v8f c = c0 + c1;

    // Epilogue: S_3 (sum of weights) lives in column 3 of the same lane-half.
    #pragma unroll
    for (int i = 0; i < 8; ++i) {
        float s3 = __shfl(c[i], (lane & 16) | 3, 32);
        if (col < 3) {
            int k = tile * 16 + (hi << 3) + i;
            out[3*k + col] = c[i] - x[3*k + col] * s3;
        }
    }
}

// ---------------------------------------------------------------------------
extern "C" void kernel_launch(void* const* d_in, const int* in_sizes, int n_in,
                              void* d_out, int out_size, void* d_ws, size_t ws_size,
                              hipStream_t stream) {
    const float* x     = (const float*)d_in[0];   // (K, 3)
    const float* means = (const float*)d_in[1];   // (M, 3)
    const float* theta = (const float*)d_in[2];   // (M,)
    float* out = (float*)d_out;                   // (K, 3)

    const int K = in_sizes[0] / 3;
    const int M = in_sizes[2];                    // 8000, multiple of 32

    // Workspace layout (M multiple of 32 -> all slices 16B aligned):
    //   [0, 10M)        f16 bh  [5][M]   (row 4 = zeros)
    //   [10M, 22M)      f32 mt  [3][M]
    //   [22M, 26M)      f32 q   [M]
    //   [26M, 30M)      f32 ths [M]
    _Float16* bh  = (_Float16*)d_ws;
    float*    mt  = (float*)((char*)d_ws + (size_t)5 * M * sizeof(_Float16));
    float*    qp  = mt + (size_t)3 * M;
    float*    ths = qp + (size_t)M;

    gauss_prep_kernel<<<(M + 255) / 256, 256, 0, stream>>>(
        means, theta, bh, mt, qp, ths, M);

    const int tiles  = (K + 15) / 16;             // 1024
    const int blocks = (tiles + 7) / 8;           // 8 waves (tiles) per block
    gauss_grad_wmma_kernel<<<blocks, 256, 0, stream>>>(
        x, bh, mt, qp, ths, out, K, M);
}
